// LayerMaskedAttention_1614907703307
// MI455X (gfx1250) — compile-verified, matrix operand order checked by reading, not run on hardware
//
#include <hip/hip_runtime.h>

#define T_SEQ   2048
#define D_HEAD  64
#define N_HEADS 16
#define N_BATCH 4
#define TK      32   // keys per tile
#define MQ      32   // queries per wave (two 16-row WMMA tiles)
#define WPB     4    // waves per block

typedef __attribute__((ext_vector_type(16))) __bf16 v16bf;
typedef __attribute__((ext_vector_type(8)))  float  v8f;

union BF16x16 { v16bf v; unsigned int u[8]; };

// f32 pair -> packed bf16 in one u32 (lo in [15:0], hi in [31:16])
// round-to-nearest via +0x8000, then one v_perm_b32
__device__ __forceinline__ unsigned int pack2_bf16(float lo, float hi) {
  unsigned int ul = __float_as_uint(lo) + 0x8000u;
  unsigned int uh = __float_as_uint(hi) + 0x8000u;
  return __builtin_amdgcn_perm(uh, ul, 0x07060302u);  // {uh[31:16], ul[31:16]}
}

// K index of the low element of VGPR pair vg for 16-bit A/B operand layout
// (lanes 0-15: V0..V3 -> K 0..7, V4..V7 -> K 16..23; lanes 16-31: +8)
__device__ __forceinline__ int kpos(int vg, int hf) {
  return (vg < 4) ? (vg * 2 + hf * 8) : (16 + (vg - 4) * 2 + hf * 8);
}

__device__ __forceinline__ v8f wmma_bf16(v16bf a, v16bf b, v8f c) {
  return __builtin_amdgcn_wmma_f32_16x16x32_bf16(
      false, a, false, b, (short)0, c, false, false);
}

// single-instruction lane-xor exchange: ds_swizzle_b32 group-of-32 mode,
// offset = {xor_mask[14:10], or_mask[9:5]=0, and_mask[4:0]=0x1F}
template <int XM>
__device__ __forceinline__ float swz_xor(float x) {
  return __int_as_float(
      __builtin_amdgcn_ds_swizzle(__float_as_int(x), (XM << 10) | 0x1F));
}

// max-reduction within a 16-lane group (xor masks < 16 stay inside the group)
__device__ __forceinline__ float rowmax16(float x) {
  x = fmaxf(x, swz_xor<1>(x));
  x = fmaxf(x, swz_xor<2>(x));
  x = fmaxf(x, swz_xor<4>(x));
  x = fmaxf(x, swz_xor<8>(x));
  return x;
}

__global__ __launch_bounds__(WPB * 32, 1) void flash_attn_causal_kernel(
    const float* __restrict__ qp, const float* __restrict__ kp,
    const float* __restrict__ vp, float* __restrict__ op) {
  __shared__ float smemV[WPB][TK * D_HEAD];  // 8 KB/wave: staged V tile
  __shared__ float smemP[WPB][MQ * TK];      // 4 KB/wave: P repack scratch

  const int lane = threadIdx.x & 31;
  const int wid  = threadIdx.x >> 5;
  const int gw   = blockIdx.x * WPB + wid;   // 0..4095
  const int qblk = gw & 63;                  // 32-query block within (b,h)
  const int bh   = gw >> 6;                  // 0..63
  const int b    = bh >> 4;
  const int h    = bh & 15;
  const int q0   = qblk * MQ;
  const int hf   = lane >> 4;
  const int l16  = lane & 15;

  // constant all-ones bf16 B operand: rowsum(P) = P x ones via WMMA
  BF16x16 bones;
  #pragma unroll
  for (int vg = 0; vg < 8; ++vg) bones.u[vg] = 0x3F803F80u;

  // ---- Q -> A operands: [query sub-tile][d chunk], 1/sqrt(64) folded in
  BF16x16 qa[2][2];
  #pragma unroll
  for (int qt = 0; qt < 2; ++qt) {
    const float* qrow = qp + ((size_t)bh * T_SEQ + q0 + qt * 16 + l16) * D_HEAD;
    float2 raw[2][8];
    #pragma unroll
    for (int c = 0; c < 2; ++c)
      #pragma unroll
      for (int vg = 0; vg < 8; ++vg)
        raw[c][vg] = *(const float2*)(qrow + kpos(vg, hf) + c * 32);
    #pragma unroll
    for (int c = 0; c < 2; ++c)
      #pragma unroll
      for (int vg = 0; vg < 8; ++vg)
        qa[qt][c].u[vg] = pack2_bf16(raw[c][vg].x * 0.125f, raw[c][vg].y * 0.125f);
  }

  // ---- flash state; row r of lane (sub-tile qt) is query q0 + qt*16 + r + 8*hf
  float m_i[2][8], l_i[2][8];
  v8f acc[2][4];
  #pragma unroll
  for (int qt = 0; qt < 2; ++qt) {
    #pragma unroll
    for (int r = 0; r < 8; ++r) { m_i[qt][r] = -1e30f; l_i[qt][r] = 0.0f; }
    #pragma unroll
    for (int t = 0; t < 4; ++t) {
      #pragma unroll
      for (int r = 0; r < 8; ++r) acc[qt][t][r] = 0.0f;
    }
  }

  float* swP = smemP[wid];
  float* swV = smemV[wid];
  const unsigned ldsV = (unsigned)(size_t)(void*)swV;  // 0-based LDS byte addr

  const int nkt = qblk + 1;  // causal limit in 32-key tiles

  for (int j = 0; j < nkt; ++j) {
    const int kb = j * TK;

    // LDS V buffer is reused across iterations; async writes are not
    // DS-ordered, so drain outstanding DS reads first.
    asm volatile("s_wait_dscnt 0" ::: "memory");

    // ---- async-stage V tile (32 keys x 64 f32 = 8 KB) into per-wave LDS:
    // 16 x global_load_async_to_lds_b128, tracked by ASYNCcnt, overlapped
    // with the K loads + S WMMAs + softmax below.
    {
      const char* gv = (const char*)(vp + ((size_t)bh * T_SEQ + kb) * D_HEAD);
      #pragma unroll
      for (int i = 0; i < 16; ++i) {
        const unsigned off = (unsigned)(lane * 16 + i * 512);
        const unsigned lds = ldsV + off;
        const unsigned long long ga = (unsigned long long)(size_t)(gv + off);
        asm volatile("global_load_async_to_lds_b128 %0, %1, off"
                     :: "v"(lds), "v"(ga) : "memory");
      }
    }

    // ---- K tile -> B operands: [key sub-tile][d chunk]; lane col = key.
    // Issue all raw loads for a sub-tile first, then pack.
    BF16x16 bk[2][2];
    #pragma unroll
    for (int t = 0; t < 2; ++t) {
      const float* krow = kp + ((size_t)bh * T_SEQ + kb + t * 16 + l16) * D_HEAD;
      float2 raw[2][8];
      #pragma unroll
      for (int c = 0; c < 2; ++c)
        #pragma unroll
        for (int vg = 0; vg < 8; ++vg)
          raw[c][vg] = *(const float2*)(krow + kpos(vg, hf) + c * 32);
      #pragma unroll
      for (int c = 0; c < 2; ++c)
        #pragma unroll
        for (int vg = 0; vg < 8; ++vg)
          bk[t][c].u[vg] = pack2_bf16(raw[c][vg].x, raw[c][vg].y);
    }

    // ---- S = (Q/8) K^T for both query sub-tiles (8 WMMAs)
    v8f s[2][2];
    #pragma unroll
    for (int qt = 0; qt < 2; ++qt) {
      #pragma unroll
      for (int t = 0; t < 2; ++t) {
        #pragma unroll
        for (int r = 0; r < 8; ++r) s[qt][t][r] = 0.0f;
        s[qt][t] = wmma_bf16(qa[qt][0].v, bk[t][0].v, s[qt][t]);
        s[qt][t] = wmma_bf16(qa[qt][1].v, bk[t][1].v, s[qt][t]);
      }
    }

    // ---- causal mask + online softmax rescale (row-sum comes from WMMA below)
    #pragma unroll
    for (int qt = 0; qt < 2; ++qt) {
      #pragma unroll
      for (int r = 0; r < 8; ++r) {
        const int qrow = q0 + qt * 16 + r + 8 * hf;
        float a0 = (kb + l16      <= qrow) ? s[qt][0][r] : -1e10f;
        float a1 = (kb + 16 + l16 <= qrow) ? s[qt][1][r] : -1e10f;
        const float mx = rowmax16(fmaxf(a0, a1));
        const float mn = fmaxf(m_i[qt][r], mx);
        const float al = __expf(m_i[qt][r] - mn);
        m_i[qt][r] = mn;
        l_i[qt][r] *= al;
        #pragma unroll
        for (int t = 0; t < 4; ++t) acc[qt][t][r] *= al;
        s[qt][0][r] = __expf(a0 - mn);
        s[qt][1][r] = __expf(a1 - mn);
      }
    }

    // ---- repack P (C layout -> A layout) through per-wave LDS
    #pragma unroll
    for (int qt = 0; qt < 2; ++qt) {
      #pragma unroll
      for (int r = 0; r < 8; ++r) {
        const int row = qt * 16 + r + 8 * hf;
        swP[row * TK + l16]      = s[qt][0][r];
        swP[row * TK + 16 + l16] = s[qt][1][r];
      }
    }
    asm volatile("" ::: "memory");
    BF16x16 pa[2];
    {
      float2 raw[2][8];
      #pragma unroll
      for (int qt = 0; qt < 2; ++qt)
        #pragma unroll
        for (int vg = 0; vg < 8; ++vg)
          raw[qt][vg] = *(const float2*)(swP + (qt * 16 + l16) * TK + kpos(vg, hf));
      #pragma unroll
      for (int qt = 0; qt < 2; ++qt)
        #pragma unroll
        for (int vg = 0; vg < 8; ++vg)
          pa[qt].u[vg] = pack2_bf16(raw[qt][vg].x, raw[qt][vg].y);
    }
    asm volatile("" ::: "memory");

    // ---- row sums via matrix unit: rs = P x ones (every column = rowsum);
    // lane's VGPR r holds the sum for row r + 8*hf — exactly l_i's indexing.
    #pragma unroll
    for (int qt = 0; qt < 2; ++qt) {
      v8f z;
      #pragma unroll
      for (int r = 0; r < 8; ++r) z[r] = 0.0f;
      const v8f rs = wmma_bf16(pa[qt].v, bones.v, z);
      #pragma unroll
      for (int r = 0; r < 8; ++r) l_i[qt][r] += rs[r];
    }

    // ---- V tile must be resident before the LDS gather
    asm volatile("s_wait_asynccnt 0" ::: "memory");

    // ---- O += P V: one shared B operand per d-tile, two PV WMMAs (8 total)
    #pragma unroll
    for (int t = 0; t < 4; ++t) {
      float rlo[8], rhi[8];
      #pragma unroll
      for (int vg = 0; vg < 8; ++vg) {
        const int kk = kpos(vg, hf);  // key offset within tile
        rlo[vg] = swV[kk * D_HEAD + t * 16 + l16];
        rhi[vg] = swV[(kk + 1) * D_HEAD + t * 16 + l16];
      }
      BF16x16 bv;
      #pragma unroll
      for (int vg = 0; vg < 8; ++vg)
        bv.u[vg] = pack2_bf16(rlo[vg], rhi[vg]);
      acc[0][t] = wmma_bf16(pa[0].v, bv.v, acc[0][t]);
      acc[1][t] = wmma_bf16(pa[1].v, bv.v, acc[1][t]);
    }
  }

  // ---- normalize and store: out[b, q, h*64 + d]
  #pragma unroll
  for (int qt = 0; qt < 2; ++qt) {
    float inv[8];
    #pragma unroll
    for (int r = 0; r < 8; ++r) inv[r] = __builtin_amdgcn_rcpf(l_i[qt][r]);
    #pragma unroll
    for (int t = 0; t < 4; ++t) {
      #pragma unroll
      for (int r = 0; r < 8; ++r) {
        const int qrow = q0 + qt * 16 + r + 8 * hf;
        op[((size_t)b * T_SEQ + qrow) * (N_HEADS * D_HEAD) + h * D_HEAD + t * 16 + l16]
            = acc[qt][t][r] * inv[r];
      }
    }
  }
}

extern "C" void kernel_launch(void* const* d_in, const int* in_sizes, int n_in,
                              void* d_out, int out_size, void* d_ws, size_t ws_size,
                              hipStream_t stream) {
  (void)in_sizes; (void)n_in; (void)out_size; (void)d_ws; (void)ws_size;
  const float* q = (const float*)d_in[0];
  const float* k = (const float*)d_in[1];
  const float* v = (const float*)d_in[2];
  float* o = (float*)d_out;
  const int total_waves = N_BATCH * N_HEADS * (T_SEQ / MQ);  // 4096
  dim3 grid(total_waves / WPB), block(WPB * 32);
  hipLaunchKernelGGL(flash_attn_causal_kernel, grid, block, 0, stream, q, k, v, o);
}
